// QuakeMamba2_79096117723883
// MI455X (gfx1250) — compile-verified, matrix operand order checked by reading, not run on hardware
//
#include <hip/hip_runtime.h>
#include <hip/hip_bf16.h>

// ---------------------------------------------------------------------------
// Mamba2 forward, last-token-only, MI455X (gfx1250, wave32, WMMA bf16).
// Fragment-ready operand layouts: every WMMA fragment is one contiguous
// 32B vector load. Weights pre-packed into [K/32][N/16][16][32] blocks so B
// fragments load directly from global, coalesced. OOB handled by index
// clamping (never-stored rows/cols may compute garbage) -> no data selects.
// GEMM A-tiles staged via gfx1250 async global->LDS copies (ASYNCcnt).
// ---------------------------------------------------------------------------

typedef __bf16 bf16_t;
typedef __attribute__((ext_vector_type(16))) __bf16 bf16x16;
typedef __attribute__((ext_vector_type(8)))  __bf16 bf16x8;
typedef __attribute__((ext_vector_type(8)))  float  floatx8;
typedef int v4i __attribute__((vector_size(16)));
typedef __attribute__((address_space(1))) v4i* as1_v4i_ptr;
typedef __attribute__((address_space(3))) v4i* as3_v4i_ptr;

#define SEQ_L   2048
#define BATCH   8
#define ROWS    (BATCH * SEQ_L)       // 16384
#define DMODEL  1024
#define DINNER  2048
#define DSTATE  128
#define NHEADS  32
#define HEADDIM 64
#define CONVDIM 2304
#define NXBCDT  2336                  // CONV_DIM + NHEADS (cols 2048..4384 of w_in)
#define QCH     64                    // scan chunk length
#define NCHUNK  (SEQ_L / QCH)         // 32

#if defined(__AMDGCN__) && __has_builtin(__builtin_amdgcn_global_load_async_to_lds_b128) && __has_builtin(__builtin_amdgcn_s_wait_asynccnt)
#define USE_ASYNC_COPY 1
#else
#define USE_ASYNC_COPY 0
#endif

static __device__ __forceinline__ bf16_t f2bf(float f) {
  unsigned u = __builtin_bit_cast(unsigned, f);
  u += 0x7FFFu + ((u >> 16) & 1u);               // round-to-nearest-even
  unsigned short h = (unsigned short)(u >> 16);
  return __builtin_bit_cast(bf16_t, h);
}
static __device__ __forceinline__ float bf2f(bf16_t b) {
  unsigned short h = __builtin_bit_cast(unsigned short, b);
  unsigned u = ((unsigned)h) << 16;
  return __builtin_bit_cast(float, u);
}
static __device__ __forceinline__ floatx8 zero_f8() {
  floatx8 z;
#pragma unroll
  for (int i = 0; i < 8; ++i) z[i] = 0.0f;
  return z;
}
static __device__ __forceinline__ bf16x8 zero_bf8() {
  bf16x8 z;
#pragma unroll
  for (int i = 0; i < 8; ++i) z[i] = f2bf(0.0f);
  return z;
}
// fragment position of K index within a 32-wide chunk (octet 1<->2 swap)
static __device__ __forceinline__ int po32(int k) {
  return (k & 7) | ((k >> 1) & 8) | ((k & 8) << 1);
}
static __device__ __forceinline__ int octswap(int o) {
  return (o == 1) ? 2 : ((o == 2) ? 1 : o);
}
static __device__ __forceinline__ int imin(int a, int b) { return a < b ? a : b; }

struct alignas(16) f4 { float x, y, z, w; };

// ---------------------------------------------------------------------------
__global__ void cvt_f32_bf16_kernel(const float* __restrict__ src,
                                    bf16_t* __restrict__ dst, int n) {
  int i = blockIdx.x * 256 + threadIdx.x;
  if (i < n) dst[i] = f2bf(src[i]);
}

// Pack f32 weight (KxN, N%16==0) into WMMA-B fragment blocks:
//   dst[ ((k/32)*NT + n/16)*512 + (n%16)*32 + po32(k%32) ]
__global__ void pack_b_kernel(const float* __restrict__ src,
                              bf16_t* __restrict__ dst, int K, int N) {
  int idx = blockIdx.x * 256 + threadIdx.x;
  if (idx >= K * N) return;
  int k = idx / N, n = idx % N;
  int NT = N >> 4;
  long o = (((long)(k >> 5) * NT + (n >> 4)) << 9) + (n & 15) * 32 + po32(k & 31);
  dst[o] = f2bf(src[idx]);
}

// ---------------------------------------------------------------------------
// GEMM: C(MxN) = A(MxK, row-major bf16) @ Bp(packed) [+bias]
// Block 64x64 tile, 4 waves; B fragments read directly from packed global.
// ---------------------------------------------------------------------------
#define TM 64
#define TN 64
#define TK 32
__global__ __launch_bounds__(128) void gemm_bf16_kernel(
    const bf16_t* __restrict__ A, long lda,
    const bf16_t* __restrict__ Bp, int ntTotal, int nt0,
    const float* __restrict__ bias,
    float* __restrict__ Cf, bf16_t* __restrict__ Cb, long ldc,
    int M, int N, int K) {
  __shared__ alignas(16) bf16_t As[TM][TK];
  const int tid  = threadIdx.x;
  const int lane = tid & 31;
  const int wave = tid >> 5;
  const int m0 = blockIdx.y * TM;
  const int n0 = blockIdx.x * TN;
  const int am    = lane & 15;
  const int khalf = lane >> 4;
  const int ktMax = (K >> 5) - 1;

  // clamped B tile indices (OOB tiles feed never-stored columns)
  int ntc[4];
#pragma unroll
  for (int t = 0; t < 4; ++t) ntc[t] = imin(nt0 + (n0 >> 4) + t, ntTotal - 1);

  floatx8 acc[4];
#pragma unroll
  for (int t = 0; t < 4; ++t) acc[t] = zero_f8();

  for (int k0 = 0; k0 < K; k0 += TK) {
    // unconditional speculative prefetch of next packed-B block
    {
      long kt1 = imin((k0 >> 5) + 1, ktMax);
      __builtin_prefetch(Bp + ((kt1 * ntTotal + ntc[wave & 3]) << 9) + lane * 16, 0, 3);
    }
    // stage A tile (clamped rows): vector loads, octet-swapped vector stores
#pragma unroll
    for (int ii = 0; ii < 2; ++ii) {
      int i = tid + ii * 128;
      int r = i >> 2, o = i & 3;
      int gm = imin(m0 + r, M - 1);
      const bf16_t* gsrc = A + (long)gm * lda + k0 + o * 8;
      bf16_t* ldst = &As[r][octswap(o) * 8];
#if USE_ASYNC_COPY
      __builtin_amdgcn_global_load_async_to_lds_b128(
          (as1_v4i_ptr)(void*)(const void*)gsrc,
          (as3_v4i_ptr)(void*)ldst, 0, 0);
#else
      *(bf16x8*)ldst = *(const bf16x8*)gsrc;
#endif
    }
#if USE_ASYNC_COPY
    __builtin_amdgcn_s_wait_asynccnt(0);
#endif
    __syncthreads();

    bf16x16 afrag = *(const bf16x16*)&As[wave * 16 + am][khalf * 16];
    const long kt = k0 >> 5;
#pragma unroll
    for (int t = 0; t < 4; ++t) {
      bf16x16 bfrag =
          *(const bf16x16*)(Bp + ((kt * ntTotal + ntc[t]) << 9) + am * 32 + khalf * 16);
      acc[t] = __builtin_amdgcn_wmma_f32_16x16x32_bf16(
          false, afrag, false, bfrag, (short)0, acc[t], false, false);
    }
    __syncthreads();
  }

  const int mh = lane >> 4;
  if (m0 + TM <= M && n0 + TN <= N) {          // fast path: full tile
#pragma unroll
    for (int t = 0; t < 4; ++t) {
      int gn = n0 + t * 16 + am;
      float bv = bias ? bias[gn] : 0.0f;
#pragma unroll
      for (int e = 0; e < 8; ++e) {
        int gm = m0 + wave * 16 + mh * 8 + e;
        float v = acc[t][e] + bv;
        if (Cf) Cf[(long)gm * ldc + gn] = v;
        if (Cb) Cb[(long)gm * ldc + gn] = f2bf(v);
      }
    }
  } else {
#pragma unroll
    for (int t = 0; t < 4; ++t) {
      int gn = n0 + t * 16 + am;
      if (gn >= N) continue;
      float bv = bias ? bias[gn] : 0.0f;
#pragma unroll
      for (int e = 0; e < 8; ++e) {
        int gm = m0 + wave * 16 + mh * 8 + e;
        if (gm >= M) continue;
        float v = acc[t][e] + bv;
        if (Cf) Cf[(long)gm * ldc + gn] = v;
        if (Cb) Cb[(long)gm * ldc + gn] = f2bf(v);
      }
    }
  }
}

// ---------------------------------------------------------------------------
// Fused conv1d+SiLU, dt softplus, log-decay. Writes:
//   dtx  (ROWS x 2048, bf16, = silu(conv) * dt_h)       [scan X operand]
//   Btg  per (b,chunk): [n(128)][pos(s)(64)] bf16       [scan A operand]
//   logdec (ROWS x 32), xh_last (8x2048), C_last (8x128)
// ---------------------------------------------------------------------------
__global__ __launch_bounds__(256) void conv_prep_kernel(
    const bf16_t* __restrict__ xbcdt,
    const float* __restrict__ conv_w, const float* __restrict__ conv_b,
    const float* __restrict__ A_log, const float* __restrict__ dt_bias,
    bf16_t* __restrict__ dtx, bf16_t* __restrict__ Btg,
    float* __restrict__ logdec, float* __restrict__ xh_last,
    float* __restrict__ C_last) {
  const int row = blockIdx.x;
  const int b = row >> 11;
  const int t = row & (SEQ_L - 1);
  const int tid = threadIdx.x;
  __shared__ float dts[NHEADS];
  if (tid < NHEADS) {
    float draw = bf2f(xbcdt[(long)row * NXBCDT + CONVDIM + tid]) + dt_bias[tid];
    float d = (draw > 20.0f) ? draw : log1pf(expf(draw));   // softplus
    dts[tid] = d;
    logdec[row * NHEADS + tid] = -expf(A_log[tid]) * d;
  }
  __syncthreads();

  for (int i = tid; i < CONVDIM / 8; i += 256) {   // 288 octets
    const int c0 = i * 8;
    bf16x8 vrow[4];
#pragma unroll
    for (int k = 0; k < 4; ++k) {
      int tt = t - 3 + k;   // uniform per block -> scalar branch
      vrow[k] = (tt >= 0) ? *(const bf16x8*)(xbcdt + (long)(row - 3 + k) * NXBCDT + c0)
                          : zero_bf8();
    }
    float av[8];
#pragma unroll
    for (int j = 0; j < 8; ++j) {
      f4 cw = *(const f4*)(conv_w + (c0 + j) * 4);
      float a = conv_b[c0 + j];
      a += bf2f(vrow[0][j]) * cw.x;
      a += bf2f(vrow[1][j]) * cw.y;
      a += bf2f(vrow[2][j]) * cw.z;
      a += bf2f(vrow[3][j]) * cw.w;
      av[j] = a / (1.0f + expf(-a));   // SiLU
    }

    if (c0 < DINNER) {
      bf16x8 ov;
      const float dtsv = dts[c0 >> 6];     // octet never crosses a head boundary
#pragma unroll
      for (int j = 0; j < 8; ++j) ov[j] = f2bf(av[j] * dtsv);
      *(bf16x8*)(dtx + (long)row * DINNER + c0) = ov;
      if (t == SEQ_L - 1) {
#pragma unroll
        for (int j = 0; j < 8; ++j) xh_last[b * DINNER + c0 + j] = av[j];
      }
    } else if (c0 < DINNER + DSTATE) {
      const int chunk = t >> 6, s = t & 63;
      const int sp = (s & 32) + po32(s & 31);
      const long blk = (long)(b * NCHUNK + chunk) * DSTATE * QCH;
#pragma unroll
      for (int j = 0; j < 8; ++j)
        Btg[blk + (long)(c0 - DINNER + j) * QCH + sp] = f2bf(av[j]);
    } else if (t == SEQ_L - 1) {
#pragma unroll
      for (int j = 0; j < 8; ++j)
        C_last[b * DSTATE + (c0 - DINNER - DSTATE) + j] = av[j];
    }
  }
}

// ---------------------------------------------------------------------------
// Chunked SSM state recurrence, one block per (batch, head), 4 waves.
// H (128x64 f32) in WMMA accumulators (2 m-tiles x 4 n-tiles per wave).
// Per chunk:  H = exp(sum logdec)*H + B_chunk^T @ (w .* dt .* x)
// Finally:    y_last = C_last . H + D * xh_last
// ---------------------------------------------------------------------------
__global__ __launch_bounds__(128) void ssd_scan_kernel(
    const bf16_t* __restrict__ dtx, const bf16_t* __restrict__ Btg,
    const float* __restrict__ logdec, const float* __restrict__ C_last,
    const float* __restrict__ xh_last, const float* __restrict__ Dp,
    float* __restrict__ Y_last) {
  const int bh = blockIdx.x;
  const int b = bh >> 5;
  const int h = bh & 31;
  const int tid  = threadIdx.x;
  const int lane = tid & 31;
  const int wave = tid >> 5;
  const int am    = lane & 15;
  const int khalf = lane >> 4;
  const int mh    = lane >> 4;

  __shared__ float  ld_s[QCH];
  __shared__ float  w_s[QCH];
  __shared__ float  cd_s;
  __shared__ alignas(16) bf16_t X_s[QCH][QCH];   // [p][pos(s)]
  __shared__ float  H_s[DSTATE][QCH];

  floatx8 acc[2][4];
#pragma unroll
  for (int i = 0; i < 2; ++i)
#pragma unroll
    for (int t = 0; t < 4; ++t) acc[i][t] = zero_f8();

  for (int c = 0; c < NCHUNK; ++c) {
    const long base_row = (long)b * SEQ_L + (long)c * QCH;
    if (tid < QCH) ld_s[tid] = logdec[(base_row + tid) * NHEADS + h];
    __syncthreads();
    if (tid < QCH) {
      float tail = 0.0f;
      for (int j = tid + 1; j < QCH; ++j) tail += ld_s[j];    // decay s -> chunk end
      w_s[tid] = expf(tail);
      if (tid == 0) cd_s = expf(tail + ld_s[0]);              // full-chunk decay
    }
    __syncthreads();

    // unconditional speculative prefetch of next chunk's A operand
    {
      int cn = imin(c + 1, NCHUNK - 1);
      __builtin_prefetch(Btg + (long)(b * NCHUNK + cn) * DSTATE * QCH + tid * 64, 0, 3);
    }

    // stage w-scaled X into fragment layout: X_s[p][pos(s)]
#pragma unroll
    for (int ii = 0; ii < 4; ++ii) {
      int i = tid + ii * 128;
      int s = i >> 3, o = i & 7;
      bf16x8 v = *(const bf16x8*)(dtx + (base_row + s) * DINNER + h * HEADDIM + o * 8);
      float ws = w_s[s];
      int sp = (s & 32) + po32(s & 31);
#pragma unroll
      for (int j = 0; j < 8; ++j) X_s[o * 8 + j][sp] = f2bf(bf2f(v[j]) * ws);
    }
    __syncthreads();

    const float cd = cd_s;
#pragma unroll
    for (int i = 0; i < 2; ++i)
#pragma unroll
      for (int t = 0; t < 4; ++t) acc[i][t] = acc[i][t] * cd;

    const bf16_t* bblk = Btg + (long)(b * NCHUNK + c) * DSTATE * QCH;
#pragma unroll
    for (int kk = 0; kk < QCH; kk += 32) {
#pragma unroll
      for (int i = 0; i < 2; ++i) {
        const int mrow = wave * 32 + i * 16 + am;
        bf16x16 afrag = *(const bf16x16*)(bblk + (long)mrow * QCH + kk + khalf * 16);
#pragma unroll
        for (int t = 0; t < 4; ++t) {
          bf16x16 bfrag = *(const bf16x16*)&X_s[t * 16 + am][kk + khalf * 16];
          acc[i][t] = __builtin_amdgcn_wmma_f32_16x16x32_bf16(
              false, afrag, false, bfrag, (short)0, acc[i][t], false, false);
        }
      }
    }
    __syncthreads();   // X_s / ld_s reused next chunk
  }

#pragma unroll
  for (int i = 0; i < 2; ++i)
#pragma unroll
    for (int t = 0; t < 4; ++t)
#pragma unroll
      for (int e = 0; e < 8; ++e)
        H_s[wave * 32 + i * 16 + mh * 8 + e][t * 16 + am] = acc[i][t][e];
  __syncthreads();

  if (tid < QCH) {
    const int p = tid;
    float sum = Dp[h] * xh_last[b * DINNER + h * HEADDIM + p];
    for (int n = 0; n < DSTATE; ++n) sum += C_last[b * DSTATE + n] * H_s[n][p];
    Y_last[b * DINNER + h * HEADDIM + p] = sum;
  }
}

// ---------------------------------------------------------------------------
__global__ __launch_bounds__(256) void gated_norm_kernel(
    const float* __restrict__ Y_last, const float* __restrict__ z_last,
    const float* __restrict__ norm_w, bf16_t* __restrict__ ynorm) {
  const int b = blockIdx.x;
  __shared__ float ybuf[DINNER];
  __shared__ float red[256];
  float part = 0.0f;
  for (int c = threadIdx.x; c < DINNER; c += 256) {
    float z = z_last[b * DINNER + c];
    float y = Y_last[b * DINNER + c] * (z / (1.0f + expf(-z)));
    ybuf[c] = y;
    part += y * y;
  }
  red[threadIdx.x] = part;
  __syncthreads();
  for (int s = 128; s > 0; s >>= 1) {
    if (threadIdx.x < s) red[threadIdx.x] += red[threadIdx.x + s];
    __syncthreads();
  }
  const float rinv = rsqrtf(red[0] / (float)DINNER + 1e-5f);
  for (int c = threadIdx.x; c < DINNER; c += 256)
    ynorm[b * DINNER + c] = f2bf(ybuf[c] * rinv * norm_w[c]);
}

__global__ void head_kernel(const float* __restrict__ yproj,
                            const float* __restrict__ w_head,
                            const float* __restrict__ b_head,
                            float* __restrict__ out) {
  const int b = blockIdx.x;
  const int j = threadIdx.x;    // 64 threads
  float sum = b_head[j];
  for (int k = 0; k < DMODEL; ++k) sum += yproj[b * DMODEL + k] * w_head[k * 64 + j];
  out[b * 64 + j] = sum;
}

// ---------------------------------------------------------------------------
extern "C" void kernel_launch(void* const* d_in, const int* in_sizes, int n_in,
                              void* d_out, int out_size, void* d_ws, size_t ws_size,
                              hipStream_t stream) {
  const float* x         = (const float*)d_in[0];
  const float* w_proj_in = (const float*)d_in[1];
  const float* b_proj_in = (const float*)d_in[2];
  const float* w_in      = (const float*)d_in[3];
  const float* conv_w    = (const float*)d_in[4];
  const float* conv_b    = (const float*)d_in[5];
  const float* A_log     = (const float*)d_in[6];
  const float* dt_bias   = (const float*)d_in[7];
  const float* Dv        = (const float*)d_in[8];
  const float* norm_w    = (const float*)d_in[9];
  const float* w_out     = (const float*)d_in[10];
  const float* w_head    = (const float*)d_in[11];
  const float* b_head    = (const float*)d_in[12];
  float* out = (float*)d_out;
  (void)in_sizes; (void)n_in; (void)out_size; (void)ws_size;

  char* base = (char*)d_ws;
  size_t off = 0;
  auto alloc = [&](size_t bytes) -> void* {
    void* p = base + off;
    off += (bytes + 255) & ~((size_t)255);
    return p;
  };
  bf16_t* x_bf     = (bf16_t*)alloc((size_t)ROWS * 64 * 2);
  bf16_t* wp_p     = (bf16_t*)alloc((size_t)64 * DMODEL * 2);        // packed
  bf16_t* win_p    = (bf16_t*)alloc((size_t)DMODEL * 4384 * 2);      // packed
  bf16_t* wout_p   = (bf16_t*)alloc((size_t)DINNER * DMODEL * 2);    // packed
  bf16_t* U_bf     = (bf16_t*)alloc((size_t)ROWS * DMODEL * 2);
  bf16_t* xbcdt_bf = (bf16_t*)alloc((size_t)ROWS * NXBCDT * 2);
  bf16_t* dtx_bf   = (bf16_t*)alloc((size_t)ROWS * DINNER * 2);
  bf16_t* Btg      = (bf16_t*)alloc((size_t)ROWS * DSTATE * 2);      // frag-ready
  float*  logdec   = (float*)alloc((size_t)ROWS * NHEADS * 4);
  float*  xh_last  = (float*)alloc((size_t)BATCH * DINNER * 4);
  float*  C_last   = (float*)alloc((size_t)BATCH * DSTATE * 4);
  float*  z_last   = (float*)alloc((size_t)BATCH * DINNER * 4);
  float*  Y_last   = (float*)alloc((size_t)BATCH * DINNER * 4);
  bf16_t* ynorm_bf = (bf16_t*)alloc((size_t)BATCH * DINNER * 2);
  float*  yproj    = (float*)alloc((size_t)BATCH * DMODEL * 4);

  cvt_f32_bf16_kernel<<<(ROWS * 64 + 255) / 256, 256, 0, stream>>>(x, x_bf, ROWS * 64);
  pack_b_kernel<<<(64 * DMODEL + 255) / 256, 256, 0, stream>>>(w_proj_in, wp_p, 64, DMODEL);
  pack_b_kernel<<<(DMODEL * 4384 + 255) / 256, 256, 0, stream>>>(w_in, win_p, DMODEL, 4384);
  pack_b_kernel<<<(DINNER * DMODEL + 255) / 256, 256, 0, stream>>>(w_out, wout_p, DINNER, DMODEL);

  auto gemm = [&](const bf16_t* A, long lda, const bf16_t* Bp, int ntTotal, int nt0,
                  const float* bias, float* Cf, bf16_t* Cb, long ldc,
                  int M, int N, int K) {
    dim3 g((N + TN - 1) / TN, (M + TM - 1) / TM);
    gemm_bf16_kernel<<<g, 128, 0, stream>>>(A, lda, Bp, ntTotal, nt0, bias, Cf, Cb,
                                            ldc, M, N, K);
  };

  // u = x @ w_proj_in + b  -> bf16
  gemm(x_bf, 64, wp_p, 64, 0, b_proj_in, nullptr, U_bf, DMODEL, ROWS, DMODEL, 64);
  // xBC|dt = u @ w_in[:, 2048:4384]
  gemm(U_bf, DMODEL, win_p, 274, 128, nullptr, nullptr, xbcdt_bf, NXBCDT,
       ROWS, NXBCDT, DMODEL);
  // z_last = u[last rows] @ w_in[:, 0:2048]  (8 strided rows of U)
  gemm(U_bf + (size_t)(SEQ_L - 1) * DMODEL, (long)SEQ_L * DMODEL, win_p, 274, 0,
       nullptr, z_last, nullptr, DINNER, BATCH, DINNER, DMODEL);

  conv_prep_kernel<<<ROWS, 256, 0, stream>>>(xbcdt_bf, conv_w, conv_b, A_log, dt_bias,
                                             dtx_bf, Btg, logdec, xh_last, C_last);

  ssd_scan_kernel<<<BATCH * NHEADS, 128, 0, stream>>>(dtx_bf, Btg, logdec, C_last,
                                                      xh_last, Dv, Y_last);

  gated_norm_kernel<<<BATCH, 256, 0, stream>>>(Y_last, z_last, norm_w, ynorm_bf);

  // yproj = ynorm @ w_out
  gemm(ynorm_bf, DINNER, wout_p, 64, 0, nullptr, yproj, nullptr, DMODEL,
       BATCH, DMODEL, DINNER);

  head_kernel<<<BATCH, 64, 0, stream>>>(yproj, w_head, b_head, out);
}
